// quanCNN_8581344657981
// MI455X (gfx1250) — 8-bit matrix operand order repaired and checked by reading, compiled, not run on hardware
//
#include <hip/hip_runtime.h>
#include <hip/hip_bf16.h>


typedef int v8i_t __attribute__((ext_vector_type(8)));

#define IMG_PER_BLOCK 32
#define THREADS 256

#if __has_builtin(__builtin_amdgcn_sudot4)
  #define USE_SUDOT4 1
#else
  #define USE_SUDOT4 0
  #pragma message("gfx1250: __builtin_amdgcn_sudot4 NOT available -- scalar dot4 fallback in use")
#endif

__device__ __forceinline__ int dot4_ss(int a, int b, int c) {
#if USE_SUDOT4
    return __builtin_amdgcn_sudot4(true, a, true, b, c, false);
#else
#pragma unroll
    for (int i = 0; i < 4; ++i)
        c += ((a << ((3 - i) * 8)) >> 24) * ((b << ((3 - i) * 8)) >> 24);
    return c;
#endif
}
__device__ __forceinline__ int dot4_us(int a, int b, int c) {
#if USE_SUDOT4
    return __builtin_amdgcn_sudot4(false, a, true, b, c, false);
#else
#pragma unroll
    for (int i = 0; i < 4; ++i)
        c += (int)(((unsigned)a >> (i * 8)) & 0xff) * ((b << ((3 - i) * 8)) >> 24);
    return c;
#endif
}

__device__ __forceinline__ float lsq_q(float x, float s, float lo, float hi) {
    float t = x / s;
    t = fminf(fmaxf(t, lo), hi);
    return rintf(t);
}

__device__ __forceinline__ int pack4(int b0, int b1, int b2, int b3) {
    return (b0 & 0xff) | ((b1 & 0xff) << 8) | ((b2 & 0xff) << 16) | ((b3 & 0xff) << 24);
}

__global__ __launch_bounds__(THREADS)
void quancnn_fused_kernel(const float* __restrict__ x,
                          const float* __restrict__ w1,
                          const float* __restrict__ w2,
                          const float* __restrict__ fc_w,
                          const float* __restrict__ fc_b,
                          const float* __restrict__ s_in_p,
                          const float* __restrict__ s_w1_p,
                          const float* __restrict__ s_a1_p,
                          const float* __restrict__ s_w2_p,
                          const float* __restrict__ s_a2_p,
                          const float* __restrict__ s_fc_p,
                          float* __restrict__ out,
                          int B) {
    __shared__ __align__(16) signed char   xq [IMG_PER_BLOCK * 784];
    __shared__ __align__(16) unsigned char a1q[IMG_PER_BLOCK * 576];
    __shared__ __align__(16) unsigned char a2q[IMG_PER_BLOCK * 256];
    __shared__ __align__(16) signed char   fcwq[16 * 256];
    __shared__ int w1pa[20];
    __shared__ int w1pb[20];
    __shared__ int w2p [96];

    const int tid = threadIdx.x;
    const int imgBase = blockIdx.x * IMG_PER_BLOCK;

    const float s_in = s_in_p[0], s_w1 = s_w1_p[0], s_a1 = s_a1_p[0];
    const float s_w2 = s_w2_p[0], s_a2 = s_a2_p[0], s_fc = s_fc_p[0];
    const float sprod1 = s_in * s_w1;
    const float sprod2 = s_a1 * s_w2;

    {
        const size_t vbase = (size_t)imgBase * 196;
        const size_t vmax  = (size_t)B * 196;
        const float4* xv = (const float4*)(x + (size_t)imgBase * 784);
        for (int i = tid; i < IMG_PER_BLOCK * 196; i += THREADS) {
            float4 v;
            if (vbase + (size_t)i < vmax) v = xv[i];
            else { v.x = 0.f; v.y = 0.f; v.z = 0.f; v.w = 0.f; }
            const int q0 = (int)lsq_q(v.x, s_in, -128.f, 127.f);
            const int q1 = (int)lsq_q(v.y, s_in, -128.f, 127.f);
            const int q2 = (int)lsq_q(v.z, s_in, -128.f, 127.f);
            const int q3 = (int)lsq_q(v.w, s_in, -128.f, 127.f);
            ((int*)xq)[i] = pack4(q0, q1, q2, q3);
        }
        if (tid < 20) {
            const int oc = tid / 5, ky = tid - oc * 5;
            int q[5];
#pragma unroll
            for (int kx = 0; kx < 5; ++kx)
                q[kx] = (int)lsq_q(w1[oc * 25 + ky * 5 + kx], s_w1, -128.f, 127.f);
            w1pa[tid] = pack4(q[0], q[1], q[2], q[3]);
            w1pb[tid] = pack4(q[4], 0, 0, 0);
        }
        if (tid < 96) {
            const int oc = tid / 12, rem = tid - oc * 12;
            const int ic = rem / 3, ky = rem - ic * 3;
            const int base = oc * 36 + ic * 9 + ky * 3;
            const int q0 = (int)lsq_q(w2[base + 0], s_w2, -128.f, 127.f);
            const int q1 = (int)lsq_q(w2[base + 1], s_w2, -128.f, 127.f);
            const int q2 = (int)lsq_q(w2[base + 2], s_w2, -128.f, 127.f);
            w2p[tid] = pack4(q0, q1, q2, 0);
        }
        for (int i = tid; i < 16 * 256; i += THREADS) {
            const int n = i >> 8, k = i & 255;
            signed char q = 0;
            if (n < 10 && k < 200)
                q = (signed char)(int)lsq_q(fc_w[n * 200 + k], s_fc, -128.f, 127.f);
            fcwq[i] = q;
        }
        for (int i = tid; i < IMG_PER_BLOCK * 256 / 4; i += THREADS)
            ((int*)a2q)[i] = 0;
    }
    __syncthreads();

    for (int task = tid; task < IMG_PER_BLOCK * 12; task += THREADS) {
        const int img = task / 12;
        const int oy  = task - img * 12;
        int r[5][7], s[5][7];
#pragma unroll
        for (int ky = 0; ky < 5; ++ky) {
            const int* rp = (const int*)(xq + img * 784 + (2 * oy + ky) * 28);
#pragma unroll
            for (int j = 0; j < 7; ++j) r[ky][j] = rp[j];
#pragma unroll
            for (int j = 0; j < 6; ++j)
                s[ky][j] = (int)(((unsigned)r[ky][j] >> 16) | ((unsigned)r[ky][j + 1] << 16));
            s[ky][6] = (int)((unsigned)r[ky][6] >> 16);
        }
#pragma unroll
        for (int oc = 0; oc < 4; ++oc) {
            int wa[5], wb[5];
#pragma unroll
            for (int ky = 0; ky < 5; ++ky) { wa[ky] = w1pa[oc * 5 + ky]; wb[ky] = w1pb[oc * 5 + ky]; }
#pragma unroll
            for (int g = 0; g < 3; ++g) {
                unsigned pd = 0;
#pragma unroll
                for (int xi = 0; xi < 4; ++xi) {
                    const int ox = g * 4 + xi;
                    const int p  = 2 * ox;
                    int acc = 0;
#pragma unroll
                    for (int ky = 0; ky < 5; ++ky) {
                        const int d1 = (p & 2) ? s[ky][p >> 2]       : r[ky][p >> 2];
                        const int d2 = (p & 2) ? s[ky][(p + 4) >> 2] : r[ky][(p + 4) >> 2];
                        acc = dot4_ss(d1, wa[ky], acc);
                        acc = dot4_ss(d2, wb[ky], acc);
                    }
                    float t = (float)acc * sprod1 / s_a1;
                    t = fminf(fmaxf(t, 0.f), 255.f);
                    pd |= ((unsigned)(int)rintf(t) & 0xffu) << (8 * xi);
                }
                *(int*)(a1q + img * 576 + oc * 144 + oy * 12 + g * 4) = (int)pd;
            }
        }
    }
    __syncthreads();

    for (int task = tid; task < IMG_PER_BLOCK * 8; task += THREADS) {
        const int img = task >> 3;
        const int oc  = task & 7;
        int acc[25];
#pragma unroll
        for (int j = 0; j < 25; ++j) acc[j] = 0;
#pragma unroll
        for (int ic = 0; ic < 4; ++ic) {
            int w[3];
#pragma unroll
            for (int ky = 0; ky < 3; ++ky) w[ky] = w2p[oc * 12 + ic * 3 + ky];
            const int* chan = (const int*)(a1q + img * 576 + ic * 144);
#pragma unroll
            for (int oy = 0; oy < 5; ++oy) {
#pragma unroll
                for (int ky = 0; ky < 3; ++ky) {
                    const int y  = 2 * oy + ky;
                    const int r0 = chan[y * 3 + 0];
                    const int r1 = chan[y * 3 + 1];
                    const int r2 = chan[y * 3 + 2];
                    const int s0 = (int)(((unsigned)r0 >> 16) | ((unsigned)r1 << 16));
                    const int s1 = (int)(((unsigned)r1 >> 16) | ((unsigned)r2 << 16));
                    const int wk = w[ky];
                    acc[oy * 5 + 0] = dot4_us(r0, wk, acc[oy * 5 + 0]);
                    acc[oy * 5 + 1] = dot4_us(s0, wk, acc[oy * 5 + 1]);
                    acc[oy * 5 + 2] = dot4_us(r1, wk, acc[oy * 5 + 2]);
                    acc[oy * 5 + 3] = dot4_us(s1, wk, acc[oy * 5 + 3]);
                    acc[oy * 5 + 4] = dot4_us(r2, wk, acc[oy * 5 + 4]);
                }
            }
        }
#pragma unroll
        for (int j = 0; j < 25; ++j) {
            float t = (float)acc[j] * sprod2 / s_a2;
            t = fminf(fmaxf(t, 0.f), 255.f);
            a2q[img * 256 + oc * 25 + j] = (unsigned char)(int)rintf(t);
        }
    }
    __syncthreads();

    if (tid < 64) {
        const int lane = tid & 31;
        const int tile = tid >> 5;
        const int hi   = lane >> 4;
        const int lo   = lane & 15;
        v8i_t c = {0, 0, 0, 0, 0, 0, 0, 0};
        const unsigned char* arow = a2q + (tile * 16 + lo) * 256;
        const signed char*   brow = fcwq + lo * 256;
#pragma unroll
        for (int kc = 0; kc < 4; ++kc) {
            v8i_t a, b;
#pragma unroll
            for (int i = 0; i < 8; ++i) {
                const int ka = kc * 64 + ((i >> 2) << 5) + (((i >> 1) & 1) << 4) + (hi << 3) + ((i & 1) << 2);
                a[i] = *(const int*)(arow + ka);
                const int kb = kc * 64 + ((i >> 2) << 5) + (((i >> 1) & 1) << 4) + (hi << 3) + ((i & 1) << 2);
                b[i] = *(const int*)(brow + kb);
            }
            c = __builtin_amdgcn_wmma_i32_16x16x64_iu8(false, a, true, b, c, false, false);
        }
        const float sfc = s_a2 * s_fc;
        const int n = lo;
#pragma unroll
        for (int r = 0; r < 8; ++r) {
            const int m = r + hi * 8;
            const int g = imgBase + tile * 16 + m;
            if (n < 10 && g < B)
                out[(size_t)g * 10 + n] = (float)c[r] * sfc + fc_b[n];
        }
    }
}

extern "C" void kernel_launch(void* const* d_in, const int* in_sizes, int n_in,
                              void* d_out, int out_size, void* d_ws, size_t ws_size,
                              hipStream_t stream) {
    const float* x    = (const float*)d_in[0];
    const float* w1   = (const float*)d_in[1];
    const float* w2   = (const float*)d_in[2];
    const float* fc_w = (const float*)d_in[3];
    const float* fc_b = (const float*)d_in[4];
    const float* s_in = (const float*)d_in[5];
    const float* s_w1 = (const float*)d_in[6];
    const float* s_a1 = (const float*)d_in[7];
    const float* s_w2 = (const float*)d_in[8];
    const float* s_a2 = (const float*)d_in[9];
    const float* s_fc = (const float*)d_in[10];
    float* out = (float*)d_out;

    const int B = in_sizes[0] / 784;
    const int nblocks = (B + IMG_PER_BLOCK - 1) / IMG_PER_BLOCK;

    quancnn_fused_kernel<<<nblocks, THREADS, 0, stream>>>(
        x, w1, w2, fc_w, fc_b, s_in, s_w1, s_a1, s_w2, s_a2, s_fc, out, B);
}
